// MS_SSIM_16312285791128
// MI455X (gfx1250) — matrix operand order repaired and checked by reading, not compiled, not run on hardware
//
#include <hip/hip_runtime.h>
#include <hip/hip_bf16.h>

typedef __attribute__((ext_vector_type(16))) _Float16 v16h;
typedef __attribute__((ext_vector_type(8)))  _Float16 v8h;
typedef __attribute__((ext_vector_type(8)))  float    v8f;

// Normalized 11-tap Gaussian, sigma = 1.5
__constant__ float c_g11[11] = {
    0.00102838f, 0.00759875f, 0.03600077f, 0.10936090f, 0.21300540f,
    0.26601160f, 0.21300540f, 0.10936090f, 0.03600077f, 0.00759875f,
    0.00102838f};

__device__ inline _Float16 gtap(int d) {
    float v = 0.0f;
    if (d >= 0 && d <= 10) v = c_g11[d];
    return (_Float16)v;
}

__device__ inline v8f wmma_f16(v16h a, v16h b, v8f c) {
    return __builtin_amdgcn_wmma_f32_16x16x32_f16(false, a, false, b,
                                                  (short)0, c, false, false);
}

// ---------------------------------------------------------------------------
// zero the per-(level,batch) accumulators: 4 levels * 16 batch * 2 values
// ---------------------------------------------------------------------------
__global__ void msssim_init_acc(float* acc) {
    int i = threadIdx.x;
    if (i < 128) acc[i] = 0.0f;
}

// ---------------------------------------------------------------------------
// fp32 -> f16 conversion of both inputs (level-0 images)
// ---------------------------------------------------------------------------
__global__ void msssim_convert(const float* __restrict__ X,
                               const float* __restrict__ Y,
                               _Float16* __restrict__ Xh,
                               _Float16* __restrict__ Yh, long n) {
    long i = (long)blockIdx.x * blockDim.x + threadIdx.x;
    if (i < n) {
        Xh[i] = (_Float16)X[i];
        Yh[i] = (_Float16)Y[i];
    }
}

// ---------------------------------------------------------------------------
// 2x2 average pool on f16 images (48 images of Hi x Wi -> Hi/2 x Wi/2)
// ---------------------------------------------------------------------------
__global__ void msssim_down(const _Float16* __restrict__ in,
                            _Float16* __restrict__ out, int Hi, int Wi,
                            long nOut) {
    long i = (long)blockIdx.x * blockDim.x + threadIdx.x;
    if (i >= nOut) return;
    int Ho = Hi >> 1, Wo = Wi >> 1;
    long imgSz = (long)Ho * Wo;
    int  img = (int)(i / imgSz);
    long rem = i % imgSz;
    int y = (int)(rem / Wo), x = (int)(rem % Wo);
    const _Float16* p = in + (long)img * Hi * Wi + (long)(2 * y) * Wi + 2 * x;
    float s = (float)p[0] + (float)p[1] + (float)p[Wi] + (float)p[Wi + 1];
    out[i] = (_Float16)(0.25f * s);
}

// ---------------------------------------------------------------------------
// One SSIM level via WMMA. One wave computes one 16x16 output tile.
//   horizontal pass: D(16x16) = A(img rows 16x32, f16) x B(banded G 32x16)
//   vertical pass:   D(16x16) = A(banded G 16x32)      x B(mid 32x16 via LDS)
// acc layout for this level: [16 batches][2]{ssim_sum, cs_sum}
// ---------------------------------------------------------------------------
__global__ __launch_bounds__(256) void msssim_level(
    const _Float16* __restrict__ Xh, const _Float16* __restrict__ Yh, int H,
    int W, int Hout, int Wout, int tx, int ty, int nimg,
    float* __restrict__ acc) {
    // transposed mid tile per wave: [col 0..15][row 0..31], stride 40 (pad)
    __shared__ __align__(16) _Float16 lds[8][16 * 40];

    const int tid  = threadIdx.x;
    const int wave = tid >> 5;
    const int lane = tid & 31;
    const bool hi  = lane >= 16;
    const int  ln  = lane & 15;

    const long tilesPerImg = (long)tx * ty;
    long wid        = (long)blockIdx.x * 8 + wave;
    const long totW = (long)nimg * tilesPerImg;
    const bool isActive = (wid < totW);
    if (!isActive) wid = 0;
    const int img = (int)(wid / tilesPerImg);
    const int t   = (int)(wid % tilesPerImg);
    const int r0  = (t / tx) * 16;
    const int c0  = (t % tx) * 16;
    const int bat = img / 3;

    const _Float16* xb = Xh + (size_t)img * H * W;
    const _Float16* yb = Yh + (size_t)img * H * W;

    // --- per-lane constant filter matrices ---------------------------------
    // B (32x16) for horizontal pass: column N = lane%16,
    // half h maps to K = h + (h>=8?8:0) + (hi?8:0).
    // A (16x32) for vertical pass: row M = lane%16,
    // half h maps to K = (hi?16:0) + h.
    v16h bfilt, afilt;
#pragma unroll
    for (int h = 0; h < 16; ++h) {
        int kB   = h + (h >= 8 ? 8 : 0) + (hi ? 8 : 0);
        bfilt[h] = gtap(kB - ln);
        int kA   = (hi ? 16 : 0) + h;
        afilt[h] = gtap(kA - ln);
    }

    // --- load the two 16-row A blocks of X and Y ---------------------------
    // A layout: row M = lane%16; halves 0..7 -> cols c0+(hi?8:0)+0..7,
    //           halves 8..15 -> cols c0+16+(hi?8:0)+0..7.
    auto loadA = [&](const _Float16* base, int rb) -> v16h {
        int r = r0 + ln + rb;
        if (r > H - 1) r = H - 1;
        int cA = c0 + (hi ? 8 : 0);
        int cB = c0 + 16 + (hi ? 8 : 0);
        if (cA > W - 8) cA = W - 8;   // only replaces fully-OOB chunks
        if (cB > W - 8) cB = W - 8;
        v8h lo  = *reinterpret_cast<const v8h*>(base + (size_t)r * W + cA);
        v8h hi8 = *reinterpret_cast<const v8h*>(base + (size_t)r * W + cB);
        v16h o;
#pragma unroll
        for (int i = 0; i < 8; ++i) {
            o[i]     = lo[i];
            o[8 + i] = hi8[i];
        }
        return o;
    };
    const v16h ax0 = loadA(xb, 0), ax1 = loadA(xb, 16);
    const v16h ay0 = loadA(yb, 0), ay1 = loadA(yb, 16);

    _Float16* myLds = &lds[wave][0];
    const int rowBase = hi ? 8 : 0;

    v8f res[5];
    const v8f cz = {0.f, 0.f, 0.f, 0.f, 0.f, 0.f, 0.f, 0.f};

#pragma unroll
    for (int q = 0; q < 5; ++q) {
        v16h a0 = ax0, a1 = ax1;
        if (q == 1) { a0 = ay0; a1 = ay1; }
#pragma unroll
        for (int i = 0; i < 16; ++i) {
            if (q == 2) { a0[i] = ax0[i] * ax0[i]; a1[i] = ax1[i] * ax1[i]; }
            if (q == 3) { a0[i] = ay0[i] * ay0[i]; a1[i] = ay1[i] * ay1[i]; }
            if (q == 4) { a0[i] = ax0[i] * ay0[i]; a1[i] = ax1[i] * ay1[i]; }
        }
        // horizontal pass: mid rows 0..15 and 16..31
        v8f m0 = wmma_f16(a0, bfilt, cz);
        v8f m1 = wmma_f16(a1, bfilt, cz);

        // stage mid transposed in LDS as f16: addr = col*40 + row
        // D layout: VGPR v -> row v + (hi?8:0), col = lane%16
        v8h s0, s1;
#pragma unroll
        for (int i = 0; i < 8; ++i) {
            s0[i] = (_Float16)m0[i];
            s1[i] = (_Float16)m1[i];
        }
        *reinterpret_cast<v8h*>(myLds + ln * 40 + rowBase)      = s0;
        *reinterpret_cast<v8h*>(myLds + ln * 40 + 16 + rowBase) = s1;

        __builtin_amdgcn_fence(__ATOMIC_ACQ_REL, "wavefront");
        __builtin_amdgcn_wave_barrier();

        // B (32x16): half h -> row K=(hi?16:0)+h, col N = lane%16
        v8h b0 = *reinterpret_cast<const v8h*>(myLds + ln * 40 + (hi ? 16 : 0));
        v8h b1 =
            *reinterpret_cast<const v8h*>(myLds + ln * 40 + (hi ? 16 : 0) + 8);
        v16h bd;
#pragma unroll
        for (int i = 0; i < 8; ++i) {
            bd[i]     = b0[i];
            bd[8 + i] = b1[i];
        }
        __builtin_amdgcn_fence(__ATOMIC_ACQ_REL, "wavefront");
        __builtin_amdgcn_wave_barrier();

        // vertical pass
        res[q] = wmma_f16(afilt, bd, cz);
    }

    // --- elementwise SSIM / CS + masked tile reduction ---------------------
    const float C1 = 0.0001f;   // (0.01*1)^2
    const float C2 = 0.0009f;   // (0.03*1)^2
    float ssim_s = 0.f, cs_s = 0.f;
    const int Mb = hi ? 8 : 0;
#pragma unroll
    for (int v = 0; v < 8; ++v) {
        float mu1 = res[0][v], mu2 = res[1][v];
        float s1  = res[2][v] - mu1 * mu1;
        float s2  = res[3][v] - mu2 * mu2;
        float s12 = res[4][v] - mu1 * mu2;
        float cs  = (2.f * s12 + C2) / (s1 + s2 + C2);
        cs        = cs > 0.f ? cs : 0.f;
        float ss  = (2.f * mu1 * mu2 + C1) / (mu1 * mu1 + mu2 * mu2 + C1) * cs;
        int r = r0 + Mb + v;
        int c = c0 + ln;
        float mask = (r < Hout && c < Wout) ? 1.f : 0.f;
        cs_s += cs * mask;
        ssim_s += ss * mask;
    }
    if (!isActive) { ssim_s = 0.f; cs_s = 0.f; }

#pragma unroll
    for (int off = 16; off >= 1; off >>= 1) {
        ssim_s += __shfl_xor(ssim_s, off, 32);
        cs_s += __shfl_xor(cs_s, off, 32);
    }
    if (lane == 0 && isActive) {
        atomicAdd(&acc[bat * 2 + 0], ssim_s);
        atomicAdd(&acc[bat * 2 + 1], cs_s);
    }
}

// ---------------------------------------------------------------------------
// combine: out[b] = 1 - prod_l max(mean_l, eps)^w_l
// ---------------------------------------------------------------------------
__global__ void msssim_finalize(const float* __restrict__ acc,
                                float* __restrict__ out) {
    int b = threadIdx.x;
    if (b >= 16) return;
    const int   Ho[4]  = {502, 246, 118, 54};
    const float wts[4] = {0.0448f, 0.2856f, 0.3001f, 0.2363f};
    const float wsum   = 0.0448f + 0.2856f + 0.3001f + 0.2363f;
    float ms = 1.f;
    for (int l = 0; l < 4; ++l) {
        float cnt = 3.0f * (float)Ho[l] * (float)Ho[l];
        float val = acc[l * 32 + b * 2 + (l == 3 ? 0 : 1)] / cnt;
        val       = fmaxf(val, 1e-8f);
        ms *= powf(val, wts[l] / wsum);
    }
    out[b] = 1.f - ms;
}

// ---------------------------------------------------------------------------
extern "C" void kernel_launch(void* const* d_in, const int* in_sizes, int n_in,
                              void* d_out, int out_size, void* d_ws,
                              size_t ws_size, hipStream_t stream) {
    const float* X = (const float*)d_in[0];
    const float* Y = (const float*)d_in[1];
    float* out     = (float*)d_out;

    char*  ws  = (char*)d_ws;
    float* acc = (float*)ws;                       // 4*16*2 floats
    _Float16* base = (_Float16*)(ws + 1024);

    const long sz[4] = {48L * 512 * 512, 48L * 256 * 256, 48L * 128 * 128,
                        48L * 64 * 64};
    _Float16* Xl[4];
    _Float16* Yl[4];
    {
        long off = 0;
        for (int l = 0; l < 4; ++l) { Xl[l] = base + off; off += sz[l]; }
        for (int l = 0; l < 4; ++l) { Yl[l] = base + off; off += sz[l]; }
    }

    msssim_init_acc<<<1, 128, 0, stream>>>(acc);

    {
        long n   = sz[0];
        int  blk = (int)((n + 255) / 256);
        msssim_convert<<<blk, 256, 0, stream>>>(X, Y, Xl[0], Yl[0], n);
    }

    const int Hs[4] = {512, 256, 128, 64};
    for (int l = 0; l < 4; ++l) {
        int H    = Hs[l];
        int Hout = H - 10;
        int tx   = (Hout + 15) / 16;
        long waves  = 48L * tx * tx;
        int  blocks = (int)((waves + 7) / 8);
        msssim_level<<<blocks, 256, 0, stream>>>(Xl[l], Yl[l], H, H, Hout,
                                                 Hout, tx, tx, 48,
                                                 acc + l * 32);
        if (l < 3) {
            long nOut = sz[l + 1];
            int  blk  = (int)((nOut + 255) / 256);
            msssim_down<<<blk, 256, 0, stream>>>(Xl[l], Xl[l + 1], H, H, nOut);
            msssim_down<<<blk, 256, 0, stream>>>(Yl[l], Yl[l + 1], H, H, nOut);
        }
    }

    msssim_finalize<<<1, 16, 0, stream>>>(acc, out);
}
